// PointNet_16956530884710
// MI455X (gfx1250) — matrix operand order repaired and checked by reading, not compiled, not run on hardware
//
#include <hip/hip_runtime.h>
#include <stdint.h>

// PointNet + ChebConv classifier for MI455X (gfx1250), wave32, WMMA f16->f32.
// All heavy GEMMs use V_WMMA_F32_16X16X32_F16. Operands are kept in
// "fragment-ready" layouts so every WMMA operand load is 2x b128:
//   A-frag (16x32):  lane = h*16 + m,  element j <-> k = 16h + j
//   B-frag (32x16):  lane = h*16 + n,  element j <-> k = 16h + j
//   C/D   (16x16):   lane n = l&15, reg r -> row m = r + 8*(l>>4)
// Requires ws_size >= ~104 MB.

typedef __attribute__((ext_vector_type(16))) _Float16 v16h;
typedef __attribute__((ext_vector_type(8)))  float    v8f;

#define B_   16
#define N_   1024
#define KNN_ 16
#define FRJ  24   // padded per-lane fragment row (halves); 48B keeps b128 align

// Load a 16-half fragment as two 128-bit loads.
__device__ __forceinline__ v16h load_frag(const _Float16* __restrict__ p) {
  v16h r;
  ((uint4*)&r)[0] = ((const uint4*)p)[0];
  ((uint4*)&r)[1] = ((const uint4*)p)[1];
  return r;
}

// Strided B-fragment read from row-major [k][n] LDS tile (activation B path).
__device__ __forceinline__ v16h frag_b_rm(const _Float16* __restrict__ base,
                                          int ld, int col0, int lane) {
  int n = lane & 15, h = lane >> 4;
  v16h b;
#pragma unroll
  for (int j = 0; j < 16; ++j) b[j] = base[(size_t)(16 * h + j) * ld + col0 + n];
  return b;
}

// A-fragment scatter map: element (k) -> (h, j)
__device__ __forceinline__ void amap(int kk, int& h, int& j) {
  h = kk >> 4;
  j = kk & 15;
}

// ---------------------------------------------------------------------------
// Small utility kernels
// ---------------------------------------------------------------------------
__global__ void k_zero_f32(float* p, int n) {
  int i = blockIdx.x * blockDim.x + threadIdx.x;
  if (i < n) p[i] = 0.f;
}

// Convert f32 weight [srcRows, N] into f16 B-fragment-swizzled layout
// [Kp/32][N/16][32][16], zero-padding rows srcRows..Kp-1.
__global__ void k_convert_swz(const float* __restrict__ src, _Float16* __restrict__ dst,
                              int srcRows, int Kp, int N) {
  int i = blockIdx.x * blockDim.x + threadIdx.x;
  if (i >= Kp * N) return;
  int k = i / N, n = i - k * N;
  int kt = k >> 5, kk = k & 31;
  int h = kk >> 4, j = kk & 15;
  int nt = n >> 4, nn = n & 15;
  size_t flat = (((size_t)kt * (N >> 4) + nt) * 32 + (h * 16 + nn)) * 16 + j;
  dst[flat] = (k < srcRows) ? (_Float16)src[(size_t)k * N + n] : (_Float16)0.f;
}

// kNN: one thread per point, register insertion sort of 16 smallest sq-dists.
__global__ void k_knn(const float* __restrict__ pos, int* __restrict__ idx) {
  int i = blockIdx.x * blockDim.x + threadIdx.x;
  if (i >= B_ * N_) return;
  int b = i >> 10;
  int pl = i & (N_ - 1);
  const float* pb = pos + (size_t)b * N_ * 3;
  float xi = pb[pl * 3 + 0], yi = pb[pl * 3 + 1], zi = pb[pl * 3 + 2];
  float bd[KNN_]; int bi[KNN_];
#pragma unroll
  for (int s = 0; s < KNN_; ++s) { bd[s] = 3.4e38f; bi[s] = 0; }
  for (int j = 0; j < N_; ++j) {
    float dx = pb[j * 3 + 0] - xi, dy = pb[j * 3 + 1] - yi, dz = pb[j * 3 + 2] - zi;
    float d = dx * dx + dy * dy + dz * dz;
    if (d < bd[KNN_ - 1]) {
      bd[KNN_ - 1] = d; bi[KNN_ - 1] = j;
#pragma unroll
      for (int s = KNN_ - 1; s > 0; --s) {
        if (bd[s] < bd[s - 1]) {
          float td = bd[s]; bd[s] = bd[s - 1]; bd[s - 1] = td;
          int   ti = bi[s]; bi[s] = bi[s - 1]; bi[s - 1] = ti;
        }
      }
    }
  }
#pragma unroll
  for (int s = 0; s < KNN_; ++s) idx[(size_t)i * KNN_ + s] = bi[s];
}

// q[i] = sum_f h[i,f]^2 over first F columns (f16 source, row stride ld)
__global__ void k_rowsq(const _Float16* __restrict__ h, int ld, int F,
                        float* __restrict__ q) {
  int i = blockIdx.x * blockDim.x + threadIdx.x;
  if (i >= B_ * N_) return;
  const _Float16* r = h + (size_t)i * ld;
  float s = 0.f;
  for (int f = 0; f < F; ++f) { float v = (float)r[f]; s += v * v; }
  q[i] = s;
}

// dinv[i] = rsqrt(sum_j A[i,j]); A (= exp adjacency) f16, row-major 1024.
__global__ void k_rowsum_dinv(const _Float16* __restrict__ A, float* __restrict__ di) {
  int i = blockIdx.x * blockDim.x + threadIdx.x;
  if (i >= B_ * N_) return;
  const _Float16* r = A + (size_t)i * N_;
  float s = 0.f;
  for (int j = 0; j < N_; ++j) s += (float)r[j];
  di[i] = rsqrtf(s);
}

// L = I - dinv_i * A * dinv_j   (in place, f16)
__global__ void k_lapfin(_Float16* __restrict__ L, const float* __restrict__ dinv) {
  long i = (long)blockIdx.x * blockDim.x + threadIdx.x;
  if (i >= (long)B_ * N_ * N_) return;
  int b = (int)(i >> 20);
  int rem = (int)(i & (N_ * N_ - 1));
  int r = rem >> 10, c = rem & (N_ - 1);
  float a = (float)L[i];
  float v = ((r == c) ? 1.f : 0.f) - dinv[b * N_ + r] * a * dinv[b * N_ + c];
  L[i] = (_Float16)v;
}

// ---------------------------------------------------------------------------
// Generic batched WMMA GEMM. C[1024, N] = A[1024, K] @ B[K, N] per batch z.
// Block tile 64(M) x 128(N); 8 waves, 2x2 WMMA fragments each; K step 32.
// BMODE: 0 = B staged row-major in LDS (activations, strided frag reads)
//        1 = B = H^T from H rows, staged fragment-ready in LDS (Laplacian)
//        2 = B read directly from pre-swizzled global weights (2x b128)
// ---------------------------------------------------------------------------
enum { EPI_STORE = 0, EPI_CHEB2, EPI_BIASRELU, EPI_LAP, EPI_THETA };

struct GemmP {
  const _Float16* A;  const _Float16* Bm;  _Float16* O;
  const _Float16* T0; const float* bias;   const float* q;
  float* gmax;
  int lda, ldb, ldo, ldt0, K;  // BMODE2: ldb = N/16 (tile count)
  long sA, sB, sO, sT0;
};

template <int EPI, int BMODE>
__global__ __launch_bounds__(256) void k_gemm(GemmP P) {
  __shared__ __align__(16) _Float16 Afr[4 * 32 * FRJ];  // 6 KB, fragment-ready
  __shared__ __align__(16) _Float16 Bsm[(BMODE == 1) ? (8 * 32 * FRJ)
                                       : (BMODE == 0) ? (32 * 136) : 16];
  const int tid  = threadIdx.x;
  const int lane = tid & 31;
  const int wave = tid >> 5;
  const int wm = (wave & 1) * 32;
  const int wn = (wave >> 1) * 32;
  const int bz = blockIdx.z;
  const int bm = blockIdx.y * 64;
  const int bn = blockIdx.x * 128;
  const _Float16* Ap = P.A  + (size_t)bz * P.sA;
  const _Float16* Bp = P.Bm + (size_t)bz * P.sB;

  v8f acc[2][2] = {};

  // A staging: thread -> (row am, 8-half chunk ac); fragment-ready dst.
  const int am = tid >> 2;
  const int ac = tid & 3;
  const int ak = ac * 8;
  _Float16* adst = &Afr[(((am >> 4) * 32) + (ac >> 1) * 16 + (am & 15)) * FRJ + (ac & 1) * 8];
  // BMODE0 staging map
  const int bkr = tid >> 3;
  const int bnc = (tid & 7) * 16;
  // BMODE1 staging map: thread -> (source row tn = output col, k chunk tkh)
  const int tn  = tid & 127;
  const int tkh = (tid >> 7) * 16;
  _Float16* bdst1 = &Bsm[(((tn >> 4) * 32) + (tkh >> 4) * 16 + (tn & 15)) * FRJ];

  for (int k0 = 0; k0 < P.K; k0 += 32) {
    *(uint4*)adst = *(const uint4*)(Ap + (size_t)(bm + am) * P.lda + k0 + ak);
    if (BMODE == 0) {
      const _Float16* src = Bp + (size_t)(k0 + bkr) * P.ldb + bn + bnc;
      *(uint4*)(Bsm + bkr * 136 + bnc)     = *(const uint4*)(src);
      *(uint4*)(Bsm + bkr * 136 + bnc + 8) = *(const uint4*)(src + 8);
    } else if (BMODE == 1) {
      const _Float16* src = Bp + (size_t)(bn + tn) * P.ldb + k0 + tkh;
      *(uint4*)(bdst1)     = *(const uint4*)(src);
      *(uint4*)(bdst1 + 8) = *(const uint4*)(src + 8);
    }
    __syncthreads();
    if (k0 + 32 < P.K)  // CDNA5 global_prefetch_b8 on the A stream
      __builtin_prefetch(Ap + (size_t)(bm + am) * P.lda + k0 + 32 + ak, 0, 1);

    v16h a0 = load_frag(&Afr[(((wm >> 4) + 0) * 32 + lane) * FRJ]);
    v16h a1 = load_frag(&Afr[(((wm >> 4) + 1) * 32 + lane) * FRJ]);
    v16h b0, b1;
    if (BMODE == 0) {
      b0 = frag_b_rm(Bsm, 136, wn + 0,  lane);
      b1 = frag_b_rm(Bsm, 136, wn + 16, lane);
    } else if (BMODE == 1) {
      b0 = load_frag(&Bsm[(((wn >> 4) + 0) * 32 + lane) * FRJ]);
      b1 = load_frag(&Bsm[(((wn >> 4) + 1) * 32 + lane) * FRJ]);
    } else {
      size_t kt = (size_t)(k0 >> 5) * P.ldb;
      b0 = load_frag(Bp + ((kt + ((bn + wn) >> 4) + 0) * 32 + lane) * 16);
      b1 = load_frag(Bp + ((kt + ((bn + wn) >> 4) + 1) * 32 + lane) * 16);
    }
    acc[0][0] = __builtin_amdgcn_wmma_f32_16x16x32_f16(false, a0, false, b0, (short)0, acc[0][0], false, false);
    acc[0][1] = __builtin_amdgcn_wmma_f32_16x16x32_f16(false, a0, false, b1, (short)0, acc[0][1], false, false);
    acc[1][0] = __builtin_amdgcn_wmma_f32_16x16x32_f16(false, a1, false, b0, (short)0, acc[1][0], false, false);
    acc[1][1] = __builtin_amdgcn_wmma_f32_16x16x32_f16(false, a1, false, b1, (short)0, acc[1][1], false, false);
    __syncthreads();
  }

  const int n = lane & 15, hh = lane >> 4;

  if (EPI == EPI_THETA) {
    // relu(acc + bias) then fused global max-pool over rows via atomicMax on
    // non-negative float bits (exact & order-independent).
#pragma unroll
    for (int fn = 0; fn < 2; ++fn) {
      int col = bn + wn + fn * 16 + n;
      float mx = -3.4e38f;
#pragma unroll
      for (int fm = 0; fm < 2; ++fm)
#pragma unroll
        for (int r = 0; r < 8; ++r) mx = fmaxf(mx, acc[fm][fn][r]);
      float v = fmaxf(mx + P.bias[col], 0.f);
      atomicMax((unsigned int*)&P.gmax[bz * N_ + col], __float_as_uint(v));
    }
    return;
  }

  _Float16* Op        = P.O  ? (P.O  + (size_t)bz * P.sO)  : (_Float16*)0;
  const _Float16* T0p = P.T0 ? (P.T0 + (size_t)bz * P.sT0) : (const _Float16*)0;
#pragma unroll
  for (int fm = 0; fm < 2; ++fm) {
#pragma unroll
    for (int fn = 0; fn < 2; ++fn) {
      int gm = bm + wm + fm * 16;
      int col = bn + wn + fn * 16 + n;
#pragma unroll
      for (int r = 0; r < 8; ++r) {
        int row = gm + r + 8 * hh;
        float v = acc[fm][fn][r];
        if (EPI == EPI_STORE) {
          Op[(size_t)row * P.ldo + col] = (_Float16)v;
        } else if (EPI == EPI_CHEB2) {
          v = 2.f * v - (float)T0p[(size_t)row * P.ldt0 + col];
          Op[(size_t)row * P.ldo + col] = (_Float16)v;
        } else if (EPI == EPI_BIASRELU) {
          v = fmaxf(v + P.bias[col], 0.f);
          Op[(size_t)row * P.ldo + col] = (_Float16)v;
        } else if (EPI == EPI_LAP) {
          float d = P.q[bz * N_ + row] + P.q[bz * N_ + col] - 2.f * v;
          Op[(size_t)row * P.ldo + col] = (_Float16)__expf(-d);
        }
      }
    }
  }
}

// ---------------------------------------------------------------------------
// Edge MLP: one wave per point (4 waves / 128-thread block).
//   inp[16, KP] = [h_j | p_j - p_i] (zero-padded K),  m1 = relu(inp@W1+b1),
//   out[point] = relu(colmax_over_16( m1@W2 + b2 ))
// W1/W2 are pre-swizzled fragment-ready; inp/m1 staged fragment-ready in LDS.
// ---------------------------------------------------------------------------
template <int KIN, int KP, bool HF16>
__global__ __launch_bounds__(128) void k_edge(const float* __restrict__ pos,
                                              const void* __restrict__ hsrc, int ldh,
                                              const int* __restrict__ idx,
                                              const _Float16* __restrict__ W1,
                                              const float* __restrict__ b1,
                                              const _Float16* __restrict__ W2,
                                              const float* __restrict__ b2,
                                              _Float16* __restrict__ outp, int ldo) {
  __shared__ __align__(16) _Float16 inp[4][(KP / 32) * 32 * FRJ];
  __shared__ __align__(16) _Float16 m1f[4][4 * 32 * FRJ];
  const int tid  = threadIdx.x;
  const int lane = tid & 31;
  const int w    = tid >> 5;
  const int pr   = blockIdx.x * 4 + w;  // global point row
  const int b    = pr >> 10;
  const int base = b * N_;
  const float* myp = pos + (size_t)(base + (pr & (N_ - 1))) * 3;
  const float px = myp[0], py = myp[1], pz = myp[2];
  const int CH = KIN - 3;
  _Float16* inpw = &inp[w][0];
  _Float16* m1w  = &m1f[w][0];

  // gather neighbors + relative positions into A-fragment layout
  for (int r = 0; r < 16; ++r) {
    int nb = idx[(size_t)pr * KNN_ + r];
    for (int c = lane; c < KP; c += 32) {
      float val;
      if (c < CH) {
        if (HF16) val = (float)((const _Float16*)hsrc)[(size_t)(base + nb) * ldh + c];
        else      val = ((const float*)hsrc)[(size_t)(base + nb) * ldh + c];
      } else if (c < KIN) {
        int d = c - CH;
        float pj = pos[(size_t)(base + nb) * 3 + d];
        float pi = (d == 0) ? px : ((d == 1) ? py : pz);
        val = pj - pi;
      } else {
        val = 0.f;
      }
      int h, j; amap(c & 31, h, j);
      inpw[(((c >> 5) * 32) + h * 16 + r) * FRJ + j] = (_Float16)val;
    }
  }

  const int n = lane & 15, hh = lane >> 4;

  // m1 = relu(inp @ W1 + b1)  [16 x 128]
#pragma unroll
  for (int nt = 0; nt < 8; ++nt) {
    v8f acc = {};
#pragma unroll
    for (int kt = 0; kt < KP / 32; ++kt) {
      v16h a  = load_frag(&inpw[(kt * 32 + lane) * FRJ]);
      v16h bb = load_frag(W1 + (((size_t)kt * 8 + nt) * 32 + lane) * 16);
      acc = __builtin_amdgcn_wmma_f32_16x16x32_f16(false, a, false, bb, (short)0, acc, false, false);
    }
    int col = nt * 16 + n;
    float bias = b1[col];
    int h, j; amap(col & 31, h, j);
#pragma unroll
    for (int r = 0; r < 8; ++r) {
      int m = r + 8 * hh;
      m1w[(((col >> 5) * 32) + h * 16 + m) * FRJ + j] =
          (_Float16)fmaxf(acc[r] + bias, 0.f);
    }
  }

  // m2 = m1 @ W2 + b2 ; column max over 16 neighbors ; relu ; store f16
#pragma unroll
  for (int nt = 0; nt < 8; ++nt) {
    v8f acc = {};
#pragma unroll
    for (int kt = 0; kt < 4; ++kt) {
      v16h a  = load_frag(&m1w[(kt * 32 + lane) * FRJ]);
      v16h bb = load_frag(W2 + (((size_t)kt * 8 + nt) * 32 + lane) * 16);
      acc = __builtin_amdgcn_wmma_f32_16x16x32_f16(false, a, false, bb, (short)0, acc, false, false);
    }
    int col = nt * 16 + n;
    float mx = -3.4e38f;
#pragma unroll
    for (int r = 0; r < 8; ++r) mx = fmaxf(mx, acc[r]);
    mx = fmaxf(mx, __shfl_xor(mx, 16, 32));  // lanes n / n+16 share column n
    mx = fmaxf(mx + b2[col], 0.f);
    if (hh == 0) outp[(size_t)pr * ldo + col] = (_Float16)mx;
  }
}

// ---------------------------------------------------------------------------
// Classifier head: 1024 -> 512 -> 128 -> 40 (tiny, fp32 VALU). One block per b.
// ---------------------------------------------------------------------------
__global__ __launch_bounds__(256) void k_classifier(const float* __restrict__ g,
    const float* __restrict__ cl1w, const float* __restrict__ cl1b,
    const float* __restrict__ cl2w, const float* __restrict__ cl2b,
    const float* __restrict__ cl3w, const float* __restrict__ cl3b,
    float* __restrict__ out) {
  __shared__ float s1[1024];
  __shared__ float s2[512];
  __shared__ float s3[128];
  int b = blockIdx.x, t = threadIdx.x;
  for (int i = t; i < 1024; i += 256) s1[i] = g[b * 1024 + i];
  __syncthreads();
  for (int o = t; o < 512; o += 256) {
    float s = cl1b[o];
    for (int f = 0; f < 1024; ++f) s += s1[f] * cl1w[f * 512 + o];
    s2[o] = fmaxf(s, 0.f);
  }
  __syncthreads();
  if (t < 128) {
    float s = cl2b[t];
    for (int f = 0; f < 512; ++f) s += s2[f] * cl2w[f * 128 + t];
    s3[t] = fmaxf(s, 0.f);
  }
  __syncthreads();
  if (t < 40) {
    float s = cl3b[t];
    for (int f = 0; f < 128; ++f) s += s3[f] * cl3w[f * 40 + t];
    out[b * 40 + t] = s;
  }
}

// ---------------------------------------------------------------------------
extern "C" void kernel_launch(void* const* d_in, const int* in_sizes, int n_in,
                              void* d_out, int out_size, void* d_ws, size_t ws_size,
                              hipStream_t stream) {
  (void)in_sizes; (void)n_in; (void)out_size; (void)ws_size;
  const float* pos  = (const float*)d_in[0];
  // d_in[1] (batch) implicit: b = row >> 10
  const float* w1a  = (const float*)d_in[2];  const float* b1a = (const float*)d_in[3];
  const float* w2a  = (const float*)d_in[4];  const float* b2a = (const float*)d_in[5];
  const float* w1b  = (const float*)d_in[6];  const float* b1b = (const float*)d_in[7];
  const float* w2b  = (const float*)d_in[8];  const float* b2b = (const float*)d_in[9];
  const float* cw1  = (const float*)d_in[10]; const float* cb1 = (const float*)d_in[11];
  const float* thet = (const float*)d_in[12]; const float* tb  = (const float*)d_in[13];
  const float* cl1w = (const float*)d_in[14]; const float* cl1b = (const float*)d_in[15];
  const float* cl2w = (const float*)d_in[16]; const float* cl2b = (const float*)d_in[17];
  const float* cl3w = (const float*)d_in[18]; const float* cl3b = (const float*)d_in[19];

  char* ws = (char*)d_ws;
  size_t cur = 0;
  auto alloc = [&](size_t bytes) -> char* {
    char* p = ws + cur;
    cur = (cur + bytes + 255) & ~(size_t)255;
    return p;
  };
  int*      idxb  = (int*)alloc((size_t)16384 * 16 * 4);
  _Float16* h1    = (_Float16*)alloc((size_t)16384 * 128 * 2);   // layer-A features
  _Float16* Tcat1 = (_Float16*)alloc((size_t)16384 * 384 * 2);   // [T0|T1|T2] @128
  _Float16* Tcat2 = (_Float16*)alloc((size_t)16384 * 1536 * 2);  // [T0|T1|T2] @512
  _Float16* Lbuf  = (_Float16*)alloc((size_t)B_ * N_ * N_ * 2);  // A then L, f16
  float*    qv    = (float*)alloc(16384 * 4);
  float*    dinv  = (float*)alloc(16384 * 4);
  float*    gbuf  = (float*)alloc(B_ * 1024 * 4);
  _Float16* w1aF  = (_Float16*)alloc(32 * 128 * 2);              // swizzled
  _Float16* w2aF  = (_Float16*)alloc(128 * 128 * 2);
  _Float16* w1bF  = (_Float16*)alloc(160 * 128 * 2);
  _Float16* w2bF  = (_Float16*)alloc(128 * 128 * 2);
  _Float16* cw1F  = (_Float16*)alloc(384 * 512 * 2);
  _Float16* thF   = (_Float16*)alloc((size_t)1536 * 1024 * 2);

  auto conv = [&](const float* s, _Float16* d, int sr, int kp, int nn) {
    int tot = kp * nn;
    k_convert_swz<<<(tot + 255) / 256, 256, 0, stream>>>(s, d, sr, kp, nn);
  };
  conv(w1a, w1aF, 6, 32, 128);
  conv(w2a, w2aF, 128, 128, 128);
  conv(w1b, w1bF, 131, 160, 128);
  conv(w2b, w2bF, 128, 128, 128);
  conv(cw1, cw1F, 384, 384, 512);
  conv(thet, thF, 1536, 1536, 1024);

  // 1) kNN
  k_knn<<<64, 256, 0, stream>>>(pos, idxb);

  // 2) PointNet layers (WMMA edge MLPs)
  k_edge<6, 32, false><<<4096, 128, 0, stream>>>(pos, pos, 3, idxb, w1aF, b1a, w2aF, b2a, h1, 128);
  k_edge<131, 160, true><<<4096, 128, 0, stream>>>(pos, h1, 128, idxb, w1bF, b1b, w2bF, b2b, Tcat1, 384);

  // 3) Laplacian 1 from h2 (Tcat1 cols 0..127): A = exp(-(qi+qj-2 H.H^T))
  k_rowsq<<<64, 256, 0, stream>>>(Tcat1, 384, 128, qv);
  {
    GemmP P{};
    P.A = Tcat1; P.Bm = Tcat1; P.O = Lbuf; P.q = qv;
    P.lda = 384; P.ldb = 384; P.ldo = 1024; P.K = 128;
    P.sA = 1024L * 384; P.sB = 1024L * 384; P.sO = 1024L * 1024;
    k_gemm<EPI_LAP, 1><<<dim3(8, 16, B_), 256, 0, stream>>>(P);
  }
  k_rowsum_dinv<<<64, 256, 0, stream>>>(Lbuf, dinv);
  k_lapfin<<<(B_ * N_ * N_) / 256, 256, 0, stream>>>(Lbuf, dinv);

  // 4) Chebyshev @128: T1 = L@T0 ; T2 = 2 L@T1 - T0
  {
    GemmP P{};
    P.A = Lbuf; P.Bm = Tcat1; P.O = Tcat1 + 128;
    P.lda = 1024; P.ldb = 384; P.ldo = 384; P.K = 1024;
    P.sA = 1024L * 1024; P.sB = 1024L * 384; P.sO = 1024L * 384;
    k_gemm<EPI_STORE, 0><<<dim3(1, 16, B_), 256, 0, stream>>>(P);
  }
  {
    GemmP P{};
    P.A = Lbuf; P.Bm = Tcat1 + 128; P.O = Tcat1 + 256; P.T0 = Tcat1;
    P.lda = 1024; P.ldb = 384; P.ldo = 384; P.ldt0 = 384; P.K = 1024;
    P.sA = 1024L * 1024; P.sB = 1024L * 384; P.sO = 1024L * 384; P.sT0 = 1024L * 384;
    k_gemm<EPI_CHEB2, 0><<<dim3(1, 16, B_), 256, 0, stream>>>(P);
  }

  // 5) h3 = relu([T0|T1|T2] @ cw1 + cb1) -> Tcat2 cols 0..511
  {
    GemmP P{};
    P.A = Tcat1; P.Bm = cw1F; P.O = Tcat2; P.bias = cb1;
    P.lda = 384; P.ldb = 512 / 16; P.ldo = 1536; P.K = 384;
    P.sA = 1024L * 384; P.sB = 0; P.sO = 1024L * 1536;
    k_gemm<EPI_BIASRELU, 2><<<dim3(4, 16, B_), 256, 0, stream>>>(P);
  }

  // 6) Laplacian 2 from h3
  k_rowsq<<<64, 256, 0, stream>>>(Tcat2, 1536, 512, qv);
  {
    GemmP P{};
    P.A = Tcat2; P.Bm = Tcat2; P.O = Lbuf; P.q = qv;
    P.lda = 1536; P.ldb = 1536; P.ldo = 1024; P.K = 512;
    P.sA = 1024L * 1536; P.sB = 1024L * 1536; P.sO = 1024L * 1024;
    k_gemm<EPI_LAP, 1><<<dim3(8, 16, B_), 256, 0, stream>>>(P);
  }
  k_rowsum_dinv<<<64, 256, 0, stream>>>(Lbuf, dinv);
  k_lapfin<<<(B_ * N_ * N_) / 256, 256, 0, stream>>>(Lbuf, dinv);

  // 7) Chebyshev @512: T1 -> cols 512..1023 ; T2 -> cols 1024..1535
  {
    GemmP P{};
    P.A = Lbuf; P.Bm = Tcat2; P.O = Tcat2 + 512;
    P.lda = 1024; P.ldb = 1536; P.ldo = 1536; P.K = 1024;
    P.sA = 1024L * 1024; P.sB = 1024L * 1536; P.sO = 1024L * 1536;
    k_gemm<EPI_STORE, 0><<<dim3(4, 16, B_), 256, 0, stream>>>(P);
  }
  {
    GemmP P{};
    P.A = Lbuf; P.Bm = Tcat2 + 512; P.O = Tcat2 + 1024; P.T0 = Tcat2;
    P.lda = 1024; P.ldb = 1536; P.ldo = 1536; P.ldt0 = 1536; P.K = 1024;
    P.sA = 1024L * 1024; P.sB = 1024L * 1536; P.sO = 1024L * 1536; P.sT0 = 1024L * 1536;
    k_gemm<EPI_CHEB2, 0><<<dim3(4, 16, B_), 256, 0, stream>>>(P);
  }

  // 8) theta einsum (= [16384,1536] @ [1536,1024]) fused with relu + max pool
  k_zero_f32<<<(B_ * 1024 + 255) / 256, 256, 0, stream>>>(gbuf, B_ * 1024);
  {
    GemmP P{};
    P.A = Tcat2; P.Bm = thF; P.bias = tb; P.gmax = gbuf;
    P.lda = 1536; P.ldb = 1024 / 16; P.K = 1536;
    P.sA = 1024L * 1536; P.sB = 0;
    k_gemm<EPI_THETA, 2><<<dim3(8, 16, B_), 256, 0, stream>>>(P);
  }

  // 9) classifier head -> d_out [16,40]
  k_classifier<<<B_, 256, 0, stream>>>(gbuf, cl1w, cl1b, cl2w, cl2b, cl3w, cl3b,
                                       (float*)d_out);
}
